// hyperbolicGCN_77266461655828
// MI455X (gfx1250) — matrix operand order repaired and checked by reading, not compiled, not run on hardware
//
#include <hip/hip_runtime.h>
#include <hip/hip_bf16.h>

// ---------------- problem constants (match reference) ----------------
#define NNODES 50000
#define DD     128
#define FIN    127
#define NE     600000
#define NLAYERS 2
#define EPSF   1e-7f
#define MINN   1e-5f
#define MAXN   1e6f
#define CLAMPF 15.0f

typedef __attribute__((ext_vector_type(16))) __bf16 v16bf;
typedef __attribute__((ext_vector_type(8)))  float  v8f;

// ---------------- small device helpers ----------------
__device__ __forceinline__ float wsum(float v) {
#pragma unroll
  for (int m = 16; m >= 1; m >>= 1) v += __shfl_xor(v, m, 32);
  return v;
}
__device__ __forceinline__ float softplusf(float x) { return log1pf(expf(x)); }
__device__ __forceinline__ float coshc(float x) { x = fminf(fmaxf(x, -CLAMPF), CLAMPF); return coshf(x); }
__device__ __forceinline__ float sinhc(float x) { x = fminf(fmaxf(x, -CLAMPF), CLAMPF); return sinhf(x); }
__device__ __forceinline__ float arcoshc(float x) { x = fmaxf(x, 1.0f + EPSF); return acoshf(x); }

// fp32 -> bf16 (round-to-nearest-even)
__device__ __forceinline__ unsigned short f2bf(float f) {
  unsigned u = __float_as_uint(f);
  unsigned r = u + 0x7fffu + ((u >> 16) & 1u);
  return (unsigned short)(r >> 16);
}

// sum over the row of v[0..3] per lane, EXCLUDING component 0 (lane0 elem0)
__device__ __forceinline__ float rnorm2_ex0(const float v[4], int lane) {
  float s = v[0]*v[0] + v[1]*v[1] + v[2]*v[2] + v[3]*v[3];
  if (lane == 0) s -= v[0]*v[0];
  return wsum(s);
}

// monotone float<->uint encoding for atomic max on floats
__device__ __forceinline__ unsigned fenc(float f) {
  unsigned u = __float_as_uint(f);
  return (u & 0x80000000u) ? ~u : (u | 0x80000000u);
}
__device__ __forceinline__ float fdec(unsigned u) {
  return (u & 0x80000000u) ? __uint_as_float(u & 0x7fffffffu) : __uint_as_float(~u);
}

// ---------------- weight prep: bf16 W^T and linW (row-major [n][k]) ----------------
// wbf layout: [l][0]=W^T, [l][1]=linW ; each 128x128 bf16
__global__ void k_prep(const float* __restrict__ Ws, const float* __restrict__ lWs,
                       unsigned short* __restrict__ wbf) {
  int tid = blockIdx.x * 256 + threadIdx.x;           // 0 .. 4*16384-1
  int mat = tid >> 14;                                // 0..3
  int idx = tid & 16383;
  int l = mat >> 1, which = mat & 1;
  int n = idx >> 7, k = idx & 127;
  float v = which ? lWs[(size_t)l * 16384 + n * 128 + k]
                  : Ws[(size_t)l * 16384 + k * 128 + n];
  wbf[tid] = f2bf(v);
}

// ---------------- h0 = proj(expmap0(proj_tan0([0,x])), c0) ----------------
__global__ void k_init(const float* __restrict__ x, const float* __restrict__ curv,
                       float* __restrict__ h) {
  int lane = threadIdx.x & 31;
  int row  = blockIdx.x * 8 + (threadIdx.x >> 5);
  if (row >= NNODES) return;
  float c = softplusf(curv[0]);
  float K = 1.0f / c, sK = sqrtf(K);
  float u[4];
#pragma unroll
  for (int j = 0; j < 4; ++j) {
    int col = 4 * lane + j;
    u[j] = (col == 0) ? 0.0f : x[(size_t)row * FIN + (col - 1)];
  }
  float xn = fmaxf(sqrtf(rnorm2_ex0(u, lane)), MINN);
  float th = xn / sK;
  float sc = sK * sinhc(th) / xn;
  float o[4];
#pragma unroll
  for (int j = 0; j < 4; ++j) o[j] = sc * u[j];
  if (lane == 0) o[0] = sK * coshc(th);
  float y2 = rnorm2_ex0(o, lane);
  if (lane == 0) o[0] = sqrtf(fmaxf(K + y2, EPSF));
  ((float4*)(h + (size_t)row * DD))[lane] = make_float4(o[0], o[1], o[2], o[3]);
}

// ---------------- t = logmap0(src, cs[l]) stored as bf16 ----------------
__global__ void k_logmap0(const float* __restrict__ src, const float* __restrict__ curv,
                          int l, unsigned short* __restrict__ t) {
  int lane = threadIdx.x & 31;
  int row  = blockIdx.x * 8 + (threadIdx.x >> 5);
  if (row >= NNODES) return;
  float c = softplusf(curv[l]);
  float sK = sqrtf(1.0f / c);
  float4 hv = ((const float4*)(src + (size_t)row * DD))[lane];
  float v[4] = {hv.x, hv.y, hv.z, hv.w};
  float yn = fmaxf(sqrtf(rnorm2_ex0(v, lane)), MINN);
  float x0 = __shfl(v[0], 0, 32);
  float th = fmaxf(x0 / sK, 1.0f + EPSF);
  float f  = sK * arcoshc(th) / yn;
  float o[4];
#pragma unroll
  for (int j = 0; j < 4; ++j) o[j] = f * v[j];
  if (lane == 0) o[0] = 0.0f;
  unsigned short* tp = t + (size_t)row * DD + 4 * lane;
  unsigned p0 = (unsigned)f2bf(o[0]) | ((unsigned)f2bf(o[1]) << 16);
  unsigned p1 = (unsigned)f2bf(o[2]) | ((unsigned)f2bf(o[3]) << 16);
  *(uint2*)tp = make_uint2(p0, p1);
}

// ---------------- bf16 WMMA GEMM: C(N x128) = A(bf16) * Bt^T ----------------
// Bt is bf16 row-major [n][k] (i.e. already the transposed effective B).
// B is staged into LDS via the gfx1250 async-to-LDS path (ASYNCcnt).
#define BSTRIDE 136
__global__ __launch_bounds__(256) void k_gemm(const unsigned short* __restrict__ A,
                                              const unsigned short* __restrict__ Bt,
                                              float* __restrict__ C) {
  __shared__ unsigned short Bl[128 * BSTRIDE];  // Bl[n][k], padded rows (272B stride)
  int tid = threadIdx.x;

  // ---- async stage: 128 rows x 256B, 16B per lane-issue ----
  unsigned ldsb = (unsigned)(size_t)(void*)&Bl[0];  // low 32 bits of flat LDS addr = LDS offset
#pragma unroll
  for (int it = 0; it < 8; ++it) {
    int idx = tid + it * 256;                 // 0..2047 chunks
    int n = idx >> 4, kc = idx & 15;
    unsigned laddr = ldsb + (unsigned)(n * (2 * BSTRIDE) + kc * 16);
    const unsigned short* gp = Bt + ((size_t)n * 128 + kc * 8);
    asm volatile("global_load_async_to_lds_b128 %0, %1, off"
                 :: "v"(laddr), "v"(gp) : "memory");
  }
  asm volatile("s_wait_asynccnt 0x0" ::: "memory");
  __syncthreads();

  int lane = tid & 31, wave = tid >> 5;
  int rowBase = blockIdx.x * 128 + wave * 16;
  if (rowBase >= NNODES) return;          // uniform per wave
  int half = lane >> 4, ml = lane & 15;
  int arow = rowBase + ml;
  if (arow >= NNODES) arow = NNODES - 1;  // clamp loads; stores are masked
  const unsigned short* Ar = A + (size_t)arow * DD;

  v8f acc[8];
#pragma unroll
  for (int nt = 0; nt < 8; ++nt) acc[nt] = (v8f){};

#pragma unroll
  for (int kt = 0; kt < 4; ++kt) {
    int koff = kt * 32;
    // A fragment (16x32 bf16): lane half selects K subset per ISA layout
    v16bf af;
#pragma unroll
    for (int i = 0; i < 8; ++i) {
      int base = ((i < 4) ? 2 * i : 16 + 2 * (i - 4)) + 8 * half;
      unsigned pv = *(const unsigned*)(Ar + koff + base);
      af[2 * i]     = __builtin_bit_cast(__bf16, (unsigned short)(pv & 0xffffu));
      af[2 * i + 1] = __builtin_bit_cast(__bf16, (unsigned short)(pv >> 16));
    }
#pragma unroll
    for (int nt = 0; nt < 8; ++nt) {
      // B fragment (32x16 bf16): same K order as A fragment (table order)
      const unsigned short* Br = Bl + (nt * 16 + ml) * BSTRIDE + koff;
      v16bf bf;
#pragma unroll
      for (int i = 0; i < 8; ++i) {
        int base = ((i < 4) ? 2 * i : 16 + 2 * (i - 4)) + 8 * half;
        unsigned pv = *(const unsigned*)(Br + base);
        bf[2 * i]     = __builtin_bit_cast(__bf16, (unsigned short)(pv & 0xffffu));
        bf[2 * i + 1] = __builtin_bit_cast(__bf16, (unsigned short)(pv >> 16));
      }
      acc[nt] = __builtin_amdgcn_wmma_f32_16x16x32_bf16(
          false, af, false, bf, (short)0, acc[nt], false, false);
    }
  }
  // store: row M = r + 8*half, col = 16*nt + (lane&15)
#pragma unroll
  for (int nt = 0; nt < 8; ++nt) {
#pragma unroll
    for (int r = 0; r < 8; ++r) {
      int row = rowBase + r + 8 * half;
      if (row < NNODES) C[(size_t)row * DD + nt * 16 + ml] = acc[nt][r];
    }
  }
}

// ---- res = proj(mobius_add(proj(expmap0(m,c)), hyp_bias(b,c), c), c) ----
__global__ void k_post_gemm1(const float* __restrict__ M, const float* __restrict__ b,
                             const float* __restrict__ curv, int l, float* __restrict__ R) {
  int lane = threadIdx.x & 31;
  int row  = blockIdx.x * 8 + (threadIdx.x >> 5);
  if (row >= NNODES) return;
  float c = softplusf(curv[l]);
  float K = 1.0f / c, sK = sqrtf(K);

  float4 mr = ((const float4*)(M + (size_t)row * DD))[lane];
  float mv[4] = {mr.x, mr.y, mr.z, mr.w};
  // expmap0(m) + proj  (component 0 of m is ignored by expmap0)
  float xn = fmaxf(sqrtf(rnorm2_ex0(mv, lane)), MINN);
  float th = xn / sK;
  float sc = sK * sinhc(th) / xn;
  float res[4];
#pragma unroll
  for (int j = 0; j < 4; ++j) res[j] = sc * mv[j];
  if (lane == 0) res[0] = sK * coshc(th);
  float y2 = rnorm2_ex0(res, lane);
  if (lane == 0) res[0] = sqrtf(fmaxf(K + y2, EPSF));

  // hyp_bias = proj(expmap0(proj_tan0(b), c))
  float bv[4];
#pragma unroll
  for (int j = 0; j < 4; ++j) {
    int col = 4 * lane + j;
    bv[j] = (col == 0) ? 0.0f : b[col];
  }
  float bn  = fmaxf(sqrtf(rnorm2_ex0(bv, lane)), MINN);
  float thb = bn / sK;
  float scb = sK * sinhc(thb) / bn;
  float hb[4];
#pragma unroll
  for (int j = 0; j < 4; ++j) hb[j] = scb * bv[j];
  if (lane == 0) hb[0] = sK * coshc(thb);
  float hb2 = rnorm2_ex0(hb, lane);
  if (lane == 0) hb[0] = sqrtf(fmaxf(K + hb2, EPSF));

  // u = logmap0(hyp_bias)
  float hbn = fmaxf(sqrtf(rnorm2_ex0(hb, lane)), MINN);
  float hb0 = __shfl(hb[0], 0, 32);
  float thu = fmaxf(hb0 / sK, 1.0f + EPSF);
  float fu  = sK * arcoshc(thu) / hbn;
  float u[4];
#pragma unroll
  for (int j = 0; j < 4; ++j) u[j] = fu * hb[j];
  if (lane == 0) u[0] = 0.0f;

  // ptransp0(res, u, c) -> w (tangent at res)
  float x0 = __shfl(res[0], 0, 32);
  float yn = fmaxf(sqrtf(rnorm2_ex0(res, lane)), MINN);
  float pd = 0.0f;
#pragma unroll
  for (int j = 0; j < 4; ++j) {
    int col = 4 * lane + j;
    if (col != 0) pd += (res[j] / yn) * u[j];
  }
  float alpha = wsum(pd) / sK;
  float w[4];
#pragma unroll
  for (int j = 0; j < 4; ++j) {
    int col = 4 * lane + j;
    float vj = (col == 0) ? (-yn) : (sK - x0) * (res[j] / yn);
    w[j] = u[j] - alpha * vj;
  }
  // proj_tan(w, res)
  float pe = 0.0f;
#pragma unroll
  for (int j = 0; j < 4; ++j) {
    int col = 4 * lane + j;
    if (col != 0) pe += res[j] * w[j];
  }
  float w0 = wsum(pe) / fmaxf(x0, EPSF);
  if (lane == 0) w[0] = w0;

  // expmap(w, res, c) + proj
  float s2 = wsum(w[0]*w[0] + w[1]*w[1] + w[2]*w[2] + w[3]*w[3]);
  float dot = s2 - 2.0f * w0 * w0;
  float normu = fminf(sqrtf(fmaxf(dot, EPSF)), MAXN);
  float thx = fmaxf(normu / sK, MINN);
  float ch = coshc(thx), sh = sinhc(thx) / thx;
  float o[4];
#pragma unroll
  for (int j = 0; j < 4; ++j) o[j] = ch * res[j] + sh * w[j];
  float o2 = rnorm2_ex0(o, lane);
  if (lane == 0) o[0] = sqrtf(fmaxf(K + o2, EPSF));
  ((float4*)(R + (size_t)row * DD))[lane] = make_float4(o[0], o[1], o[2], o[3]);
}

// ---------------- a = rowsum(hf) ----------------
__global__ void k_rowsum(const float* __restrict__ hf, float* __restrict__ a) {
  int lane = threadIdx.x & 31;
  int row  = blockIdx.x * 8 + (threadIdx.x >> 5);
  if (row >= NNODES) return;
  float4 v = ((const float4*)(hf + (size_t)row * DD))[lane];
  float s = wsum(v.x + v.y + v.z + v.w);
  if (lane == 0) a[row] = s;
}

// ---------------- zero agg / init amax,den ----------------
__global__ void k_zero_init(float* __restrict__ agg, unsigned* __restrict__ amax,
                            float* __restrict__ den) {
  int tid = blockIdx.x * 256 + threadIdx.x;
  if (tid < NNODES * DD) agg[tid] = 0.0f;
  if (tid < NNODES) { amax[tid] = 0u; den[tid] = 0.0f; }
}

// ---------------- edge pass 1: ae + segment max ----------------
__global__ void k_edge_max(const int* __restrict__ ei, const float* __restrict__ a,
                           unsigned* __restrict__ amax, float* __restrict__ ae) {
  int idx = blockIdx.x * 256 + threadIdx.x;
  if (idx >= NE + NNODES) return;
  int src = (idx < NE) ? ei[idx]      : (idx - NE);
  int dst = (idx < NE) ? ei[NE + idx] : (idx - NE);
  float v = a[src] + a[dst];
  ae[idx] = v;
  atomicMax(&amax[dst], fenc(v));
}

// ---------------- edge pass 2: exp + segment sum ----------------
__global__ void k_edge_expsum(const int* __restrict__ ei, float* __restrict__ ex,
                              const unsigned* __restrict__ amax, float* __restrict__ den) {
  int idx = blockIdx.x * 256 + threadIdx.x;
  if (idx >= NE + NNODES) return;
  int dst = (idx < NE) ? ei[NE + idx] : (idx - NE);
  float e = expf(ex[idx] - fdec(amax[dst]));
  ex[idx] = e;
  atomicAdd(&den[dst], e);
}

// ---------------- edge pass 3: agg[dst] += alpha * hf[src] ----------------
__global__ void k_edge_agg(const int* __restrict__ ei, const float* __restrict__ ex,
                           const float* __restrict__ den, const float* __restrict__ hf,
                           float* __restrict__ agg) {
  int lane = threadIdx.x & 31;
  int idx  = blockIdx.x * 8 + (threadIdx.x >> 5);   // one wave per edge
  if (idx >= NE + NNODES) return;
  int src = (idx < NE) ? ei[idx]      : (idx - NE);
  int dst = (idx < NE) ? ei[NE + idx] : (idx - NE);
  float wgt = ex[idx] / (den[dst] + 1e-16f);
  float4 hv = ((const float4*)(hf + (size_t)src * DD))[lane];
  float* ap = agg + (size_t)dst * DD + 4 * lane;
  atomicAdd(ap + 0, wgt * hv.x);
  atomicAdd(ap + 1, wgt * hv.y);
  atomicAdd(ap + 2, wgt * hv.z);
  atomicAdd(ap + 3, wgt * hv.w);
}

// ---- h = proj(expmap0(relu(logmap0(proj(expmap(res,agg,cin)),cin)),cout)) ----
__global__ void k_post_agg(const float* __restrict__ R, const float* __restrict__ G,
                           const float* __restrict__ curv, int l, float* __restrict__ H) {
  int lane = threadIdx.x & 31;
  int row  = blockIdx.x * 8 + (threadIdx.x >> 5);
  if (row >= NNODES) return;
  float cin = softplusf(curv[l]), cout = softplusf(curv[l + 1]);
  float Kin = 1.0f / cin, sKin = sqrtf(Kin);
  float Kout = 1.0f / cout, sKout = sqrtf(Kout);

  float4 rr = ((const float4*)(R + (size_t)row * DD))[lane];
  float4 gg = ((const float4*)(G + (size_t)row * DD))[lane];
  float r[4] = {rr.x, rr.y, rr.z, rr.w};
  float u[4] = {gg.x, gg.y, gg.z, gg.w};

  // expmap(u, r, cin) + proj
  float u0 = __shfl(u[0], 0, 32);
  float s2 = wsum(u[0]*u[0] + u[1]*u[1] + u[2]*u[2] + u[3]*u[3]);
  float dot = s2 - 2.0f * u0 * u0;
  float normu = fminf(sqrtf(fmaxf(dot, EPSF)), MAXN);
  float th = fmaxf(normu / sKin, MINN);
  float ch = coshc(th), sh = sinhc(th) / th;
  float z[4];
#pragma unroll
  for (int j = 0; j < 4; ++j) z[j] = ch * r[j] + sh * u[j];
  float z2 = rnorm2_ex0(z, lane);
  if (lane == 0) z[0] = sqrtf(fmaxf(Kin + z2, EPSF));

  // relu(logmap0(z, cin)), proj_tan0
  float zn = fmaxf(sqrtf(rnorm2_ex0(z, lane)), MINN);
  float z0 = __shfl(z[0], 0, 32);
  float tt = fmaxf(z0 / sKin, 1.0f + EPSF);
  float f  = sKin * arcoshc(tt) / zn;
  float t[4];
#pragma unroll
  for (int j = 0; j < 4; ++j) t[j] = fmaxf(f * z[j], 0.0f);
  if (lane == 0) t[0] = 0.0f;

  // expmap0(t, cout) + proj
  float tn = fmaxf(sqrtf(rnorm2_ex0(t, lane)), MINN);
  float th2 = tn / sKout;
  float sc = sKout * sinhc(th2) / tn;
  float o[4];
#pragma unroll
  for (int j = 0; j < 4; ++j) o[j] = sc * t[j];
  if (lane == 0) o[0] = sKout * coshc(th2);
  float o2 = rnorm2_ex0(o, lane);
  if (lane == 0) o[0] = sqrtf(fmaxf(Kout + o2, EPSF));
  ((float4*)(H + (size_t)row * DD))[lane] = make_float4(o[0], o[1], o[2], o[3]);
}

// ---------------- out = proj_tan0(logmap0(h, cs[L])) ----------------
__global__ void k_final(const float* __restrict__ h, const float* __restrict__ curv,
                        int l, float* __restrict__ out) {
  int lane = threadIdx.x & 31;
  int row  = blockIdx.x * 8 + (threadIdx.x >> 5);
  if (row >= NNODES) return;
  float c = softplusf(curv[l]);
  float sK = sqrtf(1.0f / c);
  float4 hv = ((const float4*)(h + (size_t)row * DD))[lane];
  float v[4] = {hv.x, hv.y, hv.z, hv.w};
  float yn = fmaxf(sqrtf(rnorm2_ex0(v, lane)), MINN);
  float x0 = __shfl(v[0], 0, 32);
  float th = fmaxf(x0 / sK, 1.0f + EPSF);
  float f  = sK * arcoshc(th) / yn;
  float o[4];
#pragma unroll
  for (int j = 0; j < 4; ++j) o[j] = f * v[j];
  if (lane == 0) o[0] = 0.0f;
  ((float4*)(out + (size_t)row * DD))[lane] = make_float4(o[0], o[1], o[2], o[3]);
}

// ---------------- host orchestration ----------------
extern "C" void kernel_launch(void* const* d_in, const int* in_sizes, int n_in,
                              void* d_out, int out_size, void* d_ws, size_t ws_size,
                              hipStream_t stream) {
  const float* x    = (const float*)d_in[0];
  const int*   ei   = (const int*)d_in[1];
  const float* Ws   = (const float*)d_in[2];
  const float* bs   = (const float*)d_in[3];
  const float* lWs  = (const float*)d_in[4];
  const float* curv = (const float*)d_in[5];
  float* out = (float*)d_out;

  char* p = (char*)d_ws;
  auto alloc = [&](size_t bytes) -> char* {
    char* r = p;
    p += (bytes + 255) & ~(size_t)255;
    return r;
  };
  const size_t ND = (size_t)NNODES * DD;
  float* h   = (float*)alloc(ND * 4);
  float* m   = (float*)alloc(ND * 4);   // gemm1 output; later reused as agg
  float* res = (float*)alloc(ND * 4);
  float* hf  = (float*)alloc(ND * 4);
  unsigned short* t   = (unsigned short*)alloc(ND * 2);
  unsigned short* wbf = (unsigned short*)alloc((size_t)NLAYERS * 2 * 16384 * 2);
  float*    a    = (float*)alloc((size_t)NNODES * 4);
  unsigned* amax = (unsigned*)alloc((size_t)NNODES * 4);
  float*    den  = (float*)alloc((size_t)NNODES * 4);
  float*    ex   = (float*)alloc((size_t)(NE + NNODES) * 4);

  dim3 blk(256);
  int gridRow   = (NNODES + 7) / 8;
  int gridGemm  = (NNODES + 127) / 128;
  int gridEdge  = (NE + NNODES + 255) / 256;
  int gridEdgeW = (NE + NNODES + 7) / 8;
  int gridZero  = (int)((ND + 255) / 256);

  k_prep<<<(NLAYERS * 2 * 16384) / 256, blk, 0, stream>>>(Ws, lWs, wbf);
  k_init<<<gridRow, blk, 0, stream>>>(x, curv, h);
  for (int l = 0; l < NLAYERS; ++l) {
    const unsigned short* Wt = wbf + (size_t)(2 * l) * 16384;
    const unsigned short* Lt = wbf + (size_t)(2 * l + 1) * 16384;
    k_logmap0<<<gridRow, blk, 0, stream>>>(h, curv, l, t);
    k_gemm<<<gridGemm, blk, 0, stream>>>(t, Wt, m);
    k_post_gemm1<<<gridRow, blk, 0, stream>>>(m, bs + (size_t)l * DD, curv, l, res);
    k_logmap0<<<gridRow, blk, 0, stream>>>(res, curv, l, t);
    k_gemm<<<gridGemm, blk, 0, stream>>>(t, Lt, hf);
    k_rowsum<<<gridRow, blk, 0, stream>>>(hf, a);
    k_zero_init<<<gridZero, blk, 0, stream>>>(m, amax, den);
    k_edge_max<<<gridEdge, blk, 0, stream>>>(ei, a, amax, ex);
    k_edge_expsum<<<gridEdge, blk, 0, stream>>>(ei, ex, amax, den);
    k_edge_agg<<<gridEdgeW, blk, 0, stream>>>(ei, ex, den, hf, m);
    k_post_agg<<<gridRow, blk, 0, stream>>>(res, m, curv, l, h);
  }
  k_final<<<gridRow, blk, 0, stream>>>(h, curv, NLAYERS, out);

  (void)in_sizes; (void)n_in; (void)out_size; (void)ws_size;
}
